// NeuralODEModel_87789131530507
// MI455X (gfx1250) — matrix operand order repaired and checked by reading, not compiled, not run on hardware
//
#include <hip/hip_runtime.h>
#include <cmath>

// MI455X / gfx1250, wave32. Compute-bound Neural ODE:
// all GEMM FLOPs through v_wmma_f32_16x16x32_f16, weights resident in LDS,
// RK45 state resident in registers as 16x16 f32 WMMA accumulator fragments.
// Round 3: invariant B-fragments (Wf1/Wf2) and se A-fragments hoisted into
// registers; only the dynamic h/act A-fragments touch LDS in the hot loop.

typedef __attribute__((ext_vector_type(16))) _Float16 v16h;
typedef __attribute__((ext_vector_type(8)))  float    v8f;

#define WGSIZE 256

__device__ __forceinline__ v16h load_frag(const _Float16* p0, const _Float16* p1) {
  v16h r;
#pragma unroll
  for (int i = 0; i < 8; ++i) { r[i] = p0[i]; r[i + 8] = p1[i]; }
  return r;
}

// Branch-free tanh: hardware v_tanh_f32 on gfx1250.
__device__ __forceinline__ float fast_tanh(float x) {
#if __has_builtin(__builtin_amdgcn_tanhf)
  return __builtin_amdgcn_tanhf(x);
#else
  float ax = fabsf(x);
  float e  = __expf(-2.f * ax);
  float t  = (1.f - e) * __builtin_amdgcn_rcpf(1.f + e);
  return copysignf(t, x);
#endif
}

// Branch-free softplus: max(x,0) + log(1 + exp(-|x|)).
__device__ __forceinline__ float softplus_f(float x) {
  return fmaxf(x, 0.f) + __logf(1.f + __expf(-fabsf(x)));
}

// ---------------------------------------------------------------- reductions
__global__ __launch_bounds__(256)
void reduce_mean_kernel(const float* __restrict__ sf, int B, float* __restrict__ scal) {
  __shared__ float s5[256], s6[256];
  float a5 = 0.f, a6 = 0.f;
  for (int i = threadIdx.x; i < B; i += 256) {
    a5 += sf[(size_t)i * 7 + 5];
    a6 += sf[(size_t)i * 7 + 6];
  }
  s5[threadIdx.x] = a5; s6[threadIdx.x] = a6;
  __syncthreads();
  for (int off = 128; off > 0; off >>= 1) {
    if (threadIdx.x < off) {
      s5[threadIdx.x] += s5[threadIdx.x + off];
      s6[threadIdx.x] += s6[threadIdx.x + off];
    }
    __syncthreads();
  }
  if (threadIdx.x == 0) {
    scal[0] = s5[0] / (float)B;   // amt
    scal[1] = s6[0] / (float)B;   // II
  }
}

// ------------------------------------------------------------ static encoder
__global__ __launch_bounds__(256)
void encoder_kernel(const float* __restrict__ sf,
                    const float* __restrict__ Wse1, const float* __restrict__ bse1,
                    const float* __restrict__ Wse2, const float* __restrict__ bse2,
                    const float* __restrict__ Winit, const float* __restrict__ binit,
                    const float* __restrict__ Wcp,  const float* __restrict__ bcp,
                    _Float16* __restrict__ se16, float* __restrict__ h0,
                    float* __restrict__ icp, int B)
{
  __shared__ float sW1[7 * 128];
  __shared__ float sB1[128];
  __shared__ float sW2[128 * 64];
  __shared__ float sB2[64];
  __shared__ float sWi[64 * 64];
  __shared__ float sBi[64];
  __shared__ float sWc[64];
  const int tid = threadIdx.x;
  for (int i = tid; i < 7 * 128; i += 256) sW1[i] = Wse1[i];
  for (int i = tid; i < 128; i += 256)     sB1[i] = bse1[i];
  for (int i = tid; i < 128 * 64; i += 256) sW2[i] = Wse2[i];
  for (int i = tid; i < 64 * 64; i += 256)  sWi[i] = Winit[i];
  if (tid < 64) { sB2[tid] = bse2[tid]; sBi[tid] = binit[tid]; sWc[tid] = Wcp[tid]; }
  __syncthreads();

  const int b = blockIdx.x * 256 + tid;
  if (b >= B) return;

  float x[7];
#pragma unroll
  for (int i = 0; i < 7; ++i) x[i] = sf[(size_t)b * 7 + i];

  float se[64];
#pragma unroll
  for (int k = 0; k < 64; ++k) se[k] = sB2[k];
  for (int j = 0; j < 128; ++j) {
    float a = sB1[j];
#pragma unroll
    for (int i = 0; i < 7; ++i) a += x[i] * sW1[i * 128 + j];
    a = fmaxf(a, 0.f);
#pragma unroll
    for (int k = 0; k < 64; ++k) se[k] += a * sW2[j * 64 + k];
  }
#pragma unroll
  for (int k = 0; k < 64; ++k) {
    se[k] = fmaxf(se[k], 0.f);
    se16[(size_t)b * 64 + k] = (_Float16)se[k];
  }
  for (int o = 0; o < 64; ++o) {
    float s = sBi[o];
#pragma unroll
    for (int k = 0; k < 64; ++k) s += se[k] * sWi[k * 64 + o];
    h0[(size_t)b * 64 + o] = s;
  }
  float s = bcp[0];
#pragma unroll
  for (int k = 0; k < 64; ++k) s += se[k] * sWc[k];
  icp[b] = softplus_f(s);
}

// -------------------------------------------------------------- ODE function
// One f-eval: D = tanh([h|se|t] @ Wf1 + bf1') @ Wf2 + bf2'  (bias/t/inj folded
// into accumulator init). 4 + 2 v_wmma_f32_16x16x32_f16 per wave; all weight
// (B) fragments and the se A-fragments live in registers.
__device__ __forceinline__ v8f f_eval(const _Float16* __restrict__ sXh,
                                      _Float16* __restrict__ sAt,
                                      const v16h (&bW1)[4], const v16h (&bW2)[2],
                                      const v16h (&aSe)[2],
                                      float cin1, float cin2,
                                      int row, int khi, int col)
{
  v8f acc;
#pragma unroll
  for (int m = 0; m < 8; ++m) acc[m] = cin1;
  // dynamic h part (K = 0..63)
#pragma unroll
  for (int kc = 0; kc < 2; ++kc) {
    const _Float16* ap = sXh + row * 64 + kc * 32 + khi * 8;
    v16h a = load_frag(ap, ap + 16);
    acc = __builtin_amdgcn_wmma_f32_16x16x32_f16(false, a, false, bW1[kc], (short)0, acc, false, false);
  }
  // static se part (K = 64..127), A-fragments in registers
#pragma unroll
  for (int kc = 0; kc < 2; ++kc)
    acc = __builtin_amdgcn_wmma_f32_16x16x32_f16(false, aSe[kc], false, bW1[2 + kc], (short)0, acc, false, false);

#pragma unroll
  for (int m = 0; m < 8; ++m)
    sAt[(m + 8 * khi) * 64 + col] = (_Float16)fast_tanh(acc[m]);
  __syncthreads();

  v8f acc2;
#pragma unroll
  for (int m = 0; m < 8; ++m) acc2[m] = cin2;
#pragma unroll
  for (int kc = 0; kc < 2; ++kc) {
    const _Float16* ap = sAt + row * 64 + kc * 32 + khi * 8;
    v16h a = load_frag(ap, ap + 16);
    acc2 = __builtin_amdgcn_wmma_f32_16x16x32_f16(false, a, false, bW2[kc], (short)0, acc2, false, false);
  }
  return acc2;
}

// ------------------------------------------------------------- ODE integrator
// Block: 256 threads = 8 waves. 32 batch rows per block (2 tiles x 4 waves).
// Wave (tile,nt) owns the 16x16 output fragment cols [nt*16, nt*16+16).
__global__ __launch_bounds__(256)
void ode_kernel(const float* __restrict__ tt,
                const _Float16* __restrict__ se16,
                const float* __restrict__ h0,
                const float* __restrict__ icp,
                const float* __restrict__ scal,
                const float* __restrict__ Wf1, const float* __restrict__ bf1,
                const float* __restrict__ Wf2, const float* __restrict__ bf2,
                const float* __restrict__ injscale_p,
                const float* __restrict__ Wro, const float* __restrict__ bro_p,
                float* __restrict__ out, int T)
{
  __shared__ _Float16 sWf1[4 * 4 * 32 * 16];  // [kchunk][ntile][lane][16] (prologue only)
  __shared__ _Float16 sWf2[2 * 4 * 32 * 16];  // (prologue only)
  __shared__ _Float16 sXh[2][16 * 64];        // [tile] stage-state h rows (f16)
  __shared__ _Float16 sAct[2][16 * 64];
  __shared__ float    sH[2][16 * 64];
  __shared__ float    sWro[64];

  const int tid = threadIdx.x;
  const int blockBase = blockIdx.x * 32;

  // Stage Wf1/Wf2 into LDS in WMMA B-fragment order (col = lane&15, K 8-blocks by lane>>4, table order).
  for (int idx = tid; idx < 4 * 4 * 32 * 16; idx += WGSIZE) {
    int i = idx & 15, l = (idx >> 4) & 31, nt = (idx >> 9) & 3, kc = idx >> 11;
    int K = kc * 32 + (i >> 3) * 16 + (l >> 4) * 8 + (i & 7), N = nt * 16 + (l & 15);
    sWf1[idx] = (_Float16)Wf1[K * 64 + N];
  }
  for (int idx = tid; idx < 2 * 4 * 32 * 16; idx += WGSIZE) {
    int i = idx & 15, l = (idx >> 4) & 31, nt = (idx >> 9) & 3, kc = idx >> 11;
    int K = kc * 32 + (i >> 3) * 16 + (l >> 4) * 8 + (i & 7), N = nt * 16 + (l & 15);
    sWf2[idx] = (_Float16)Wf2[K * 64 + N];
  }
  if (tid < 64) sWro[tid] = Wro[tid];
  __syncthreads();

  const int lane = tid & 31;
  const int wave = tid >> 5;
  const int tile = wave >> 2;
  const int nt   = wave & 3;
  const int row  = lane & 15;       // A-fragment row
  const int khi  = lane >> 4;       // K/M half select
  const int col  = nt * 16 + (lane & 15);  // C/D fragment column
  const int gRow0 = blockBase + tile * 16;

  _Float16* sXt = sXh[tile];
  _Float16* sAt = sAct[tile];

  // Hoist invariant fragments into registers.
  v16h bW1[4], bW2[2], aSe[2];
#pragma unroll
  for (int kc = 0; kc < 4; ++kc) {
    const _Float16* bp = sWf1 + nt * (32 * 16) + kc * (4 * 32 * 16) + lane * 16;
    bW1[kc] = load_frag(bp, bp + 8);
  }
#pragma unroll
  for (int kc = 0; kc < 2; ++kc) {
    const _Float16* bp = sWf2 + nt * (32 * 16) + kc * (4 * 32 * 16) + lane * 16;
    bW2[kc] = load_frag(bp, bp + 8);
  }
  // se A-fragments straight from global (row-contiguous -> b128 loads).
#pragma unroll
  for (int kc = 0; kc < 2; ++kc) {
    const _Float16* gp = se16 + (size_t)(gRow0 + row) * 64 + kc * 32 + khi * 8;
    aSe[kc] = load_frag(gp, gp + 16);
  }

  const float amt   = scal[0];
  const float II    = scal[1];
  const float injsc = injscale_p[0];
  const float bf1c  = bf1[col];
  const float w128c = Wf1[128 * 64 + col];  // t-column of Wf1 folded into C-init
  const float bf2c  = bf2[col];
  const float bro   = bro_p[0];
  const float INV2W2 = 5.0e5f;              // 1/(2*W_INJ^2), W_INJ = 1e-3

  auto cin1 = [&](float ts) { return bf1c + ts * w128c; };
  auto cin2 = [&](float ts) {
    float rem = fmodf(ts, II);
    return bf2c + injsc * amt * __expf(-rem * rem * INV2W2);
  };

  // h state as a 16x16 f32 C/D fragment per wave.
  v8f h;
#pragma unroll
  for (int m = 0; m < 8; ++m)
    h[m] = h0[(size_t)(gRow0 + khi * 8 + m) * 64 + col];

  auto do_readout = [&](int tstep) {
#pragma unroll
    for (int m = 0; m < 8; ++m) sH[tile][(m + 8 * khi) * 64 + col] = h[m];
    __syncthreads();
    if (tid < 32) {
      int tl = tid >> 4, r = tid & 15;
      int gb = blockBase + tl * 16 + r;
      float s = bro;
      for (int c = 0; c < 64; ++c) s += sH[tl][r * 64 + c] * sWro[c];
      out[(size_t)gb * T + tstep] = softplus_f(s) + icp[gb];
    }
    __syncthreads();
  };

  do_readout(0);

  for (int it = 0; it < T - 1; ++it) {
    const float t0 = tt[it];
    const float dt = tt[it + 1] - t0;

    // ---- stage 1
#pragma unroll
    for (int m = 0; m < 8; ++m)
      sXt[(m + 8 * khi) * 64 + col] = (_Float16)h[m];
    __syncthreads();
    v8f k1 = f_eval(sXt, sAt, bW1, bW2, aSe, cin1(t0), cin2(t0), row, khi, col);

    // ---- stage 2
    float ts = t0 + 0.2f * dt;
#pragma unroll
    for (int m = 0; m < 8; ++m)
      sXt[(m + 8 * khi) * 64 + col] = (_Float16)(h[m] + dt * (0.2f * k1[m]));
    __syncthreads();
    v8f k2 = f_eval(sXt, sAt, bW1, bW2, aSe, cin1(ts), cin2(ts), row, khi, col);

    // ---- stage 3
    ts = t0 + 0.3f * dt;
#pragma unroll
    for (int m = 0; m < 8; ++m)
      sXt[(m + 8 * khi) * 64 + col] =
          (_Float16)(h[m] + dt * (0.075f * k1[m] + 0.225f * k2[m]));
    __syncthreads();
    v8f k3 = f_eval(sXt, sAt, bW1, bW2, aSe, cin1(ts), cin2(ts), row, khi, col);

    // ---- stage 4
    ts = t0 + 0.8f * dt;
#pragma unroll
    for (int m = 0; m < 8; ++m)
      sXt[(m + 8 * khi) * 64 + col] =
          (_Float16)(h[m] + dt * ((44.f / 45.f) * k1[m] + (-56.f / 15.f) * k2[m] +
                                  (32.f / 9.f) * k3[m]));
    __syncthreads();
    v8f k4 = f_eval(sXt, sAt, bW1, bW2, aSe, cin1(ts), cin2(ts), row, khi, col);

    // ---- stage 5
    ts = t0 + (8.f / 9.f) * dt;
#pragma unroll
    for (int m = 0; m < 8; ++m)
      sXt[(m + 8 * khi) * 64 + col] =
          (_Float16)(h[m] + dt * ((19372.f / 6561.f) * k1[m] + (-25360.f / 2187.f) * k2[m] +
                                  (64448.f / 6561.f) * k3[m] + (-212.f / 729.f) * k4[m]));
    __syncthreads();
    v8f k5 = f_eval(sXt, sAt, bW1, bW2, aSe, cin1(ts), cin2(ts), row, khi, col);

    // ---- stage 6
    ts = t0 + dt;
#pragma unroll
    for (int m = 0; m < 8; ++m)
      sXt[(m + 8 * khi) * 64 + col] =
          (_Float16)(h[m] + dt * ((9017.f / 3168.f) * k1[m] + (-355.f / 33.f) * k2[m] +
                                  (46732.f / 5247.f) * k3[m] + (49.f / 176.f) * k4[m] +
                                  (-5103.f / 18656.f) * k5[m]));
    __syncthreads();
    v8f k6 = f_eval(sXt, sAt, bW1, bW2, aSe, cin1(ts), cin2(ts), row, khi, col);

    // ---- 5th-order combination
#pragma unroll
    for (int m = 0; m < 8; ++m)
      h[m] += dt * ((35.f / 384.f) * k1[m] + (500.f / 1113.f) * k3[m] +
                    (125.f / 192.f) * k4[m] + (-2187.f / 6784.f) * k5[m] +
                    (11.f / 84.f) * k6[m]);

    do_readout(it + 1);
  }
}

// ------------------------------------------------------------------- launch
extern "C" void kernel_launch(void* const* d_in, const int* in_sizes, int n_in,
                              void* d_out, int out_size, void* d_ws, size_t ws_size,
                              hipStream_t stream) {
  const float* t     = (const float*)d_in[0];
  const float* sf    = (const float*)d_in[1];
  const float* Wse1  = (const float*)d_in[2];
  const float* bse1  = (const float*)d_in[3];
  const float* Wse2  = (const float*)d_in[4];
  const float* bse2  = (const float*)d_in[5];
  const float* Winit = (const float*)d_in[6];
  const float* binit = (const float*)d_in[7];
  const float* Wf1   = (const float*)d_in[8];
  const float* bf1   = (const float*)d_in[9];
  const float* Wf2   = (const float*)d_in[10];
  const float* bf2   = (const float*)d_in[11];
  const float* injs  = (const float*)d_in[12];
  const float* Wro   = (const float*)d_in[13];
  const float* bro   = (const float*)d_in[14];
  const float* Wcp   = (const float*)d_in[15];
  const float* bcp   = (const float*)d_in[16];

  const int T = in_sizes[0];
  const int B = in_sizes[1] / 7;

  // workspace carve-up (~3.1 MB for B=8192)
  char* w = (char*)d_ws;
  float*    scal = (float*)w;                                    // 2 scalars
  _Float16* se16 = (_Float16*)(w + 256);                         // B*64 f16
  float*    h0   = (float*)(w + 256 + (size_t)B * 64 * 2);       // B*64 f32
  float*    icp  = (float*)(w + 256 + (size_t)B * 64 * 2 + (size_t)B * 64 * 4);  // B f32

  reduce_mean_kernel<<<1, 256, 0, stream>>>(sf, B, scal);
  encoder_kernel<<<(B + 255) / 256, 256, 0, stream>>>(
      sf, Wse1, bse1, Wse2, bse2, Winit, binit, Wcp, bcp, se16, h0, icp, B);
  ode_kernel<<<B / 32, 256, 0, stream>>>(
      t, se16, h0, icp, scal, Wf1, bf1, Wf2, bf2, injs, Wro, bro, (float*)d_out, T);
}
